// cvx_56006373540381
// MI455X (gfx1250) — matrix operand order repaired and checked by reading, not compiled, not run on hardware
//
#include <hip/hip_runtime.h>

// ---------------------------------------------------------------------------
// Problem constants (from the reference)
// ---------------------------------------------------------------------------
#define NN    3500      // nodes
#define EE    4000      // edges
#define DIN   128
#define HH    1024
#define LL    512
#define MPAD  3584      // NN padded to multiple of 128
#define VLO   0.9f
#define VUP   1.1f

// output layout: yw[0,4000) vw[4000,7500) y_un[7500,11500) sqrtv[11500,15000)
#define OFF_YW  0
#define OFF_VW  4000
#define OFF_YU  7500
#define OFF_SV  11500

// ---------------------------------------------------------------------------
// WMMA types (gfx1250, wave32)
// ---------------------------------------------------------------------------
typedef __attribute__((ext_vector_type(16))) __bf16 v16bf;
typedef __attribute__((ext_vector_type(8)))  float  v8f;
typedef int v4i __attribute__((vector_size(16)));

__device__ __forceinline__ unsigned short f2bf(float f) {
    union { float f; unsigned u; } v; v.f = f;
    unsigned u = v.u;
    unsigned r = u + 0x7FFFu + ((u >> 16) & 1u);   // RNE fp32 -> bf16
    return (unsigned short)(r >> 16);
}

// ---------------------------------------------------------------------------
// Optional CDNA5 async global->LDS staging (ASYNCcnt path)
//   prototype (from probe diagnostic): (v4i32 AS1* src, v4i32 AS3* dst,
//                                       imm int offset, imm int cpol)
// ---------------------------------------------------------------------------
#if defined(__has_builtin)
#  if __has_builtin(__builtin_amdgcn_global_load_async_to_lds_b128)
#    define USE_ASYNC_LDS 1
#  endif
#endif
#ifndef USE_ASYNC_LDS
#  define USE_ASYNC_LDS 0
#endif

#if USE_ASYNC_LDS
typedef __attribute__((address_space(1))) v4i* gv4i_p;
typedef __attribute__((address_space(3))) v4i* lv4i_p;

__device__ __forceinline__ void async_b128(const unsigned short* g, unsigned short* l) {
    __builtin_amdgcn_global_load_async_to_lds_b128(
        (gv4i_p)(void*)g, (lv4i_p)(void*)l, 0, 0);
}
__device__ __forceinline__ void wait_async0() {
#  if __has_builtin(__builtin_amdgcn_s_wait_asynccnt)
    __builtin_amdgcn_s_wait_asynccnt(0);
#  else
    asm volatile("s_wait_asynccnt 0x0" ::: "memory");
#  endif
}
#endif

// ---------------------------------------------------------------------------
// Casting / transpose kernels
// ---------------------------------------------------------------------------
__global__ void cast_x_kernel(const float* __restrict__ x, unsigned short* __restrict__ xb) {
    int i = blockIdx.x * blockDim.x + threadIdx.x;
    if (i >= MPAD * DIN) return;
    int r = i / DIN;
    float v = (r < NN) ? x[i] : 0.0f;   // zero-pad rows [NN, MPAD)
    xb[i] = f2bf(v);
}

// W [K][Nc] fp32 (row-major)  ->  Wt [Nc][K] bf16
__global__ void castT_w_kernel(const float* __restrict__ W, unsigned short* __restrict__ Wt,
                               int K, int Nc) {
    int i = blockIdx.x * blockDim.x + threadIdx.x;
    if (i >= K * Nc) return;
    int k = i / Nc, n = i - k * Nc;
    Wt[n * K + k] = f2bf(W[i]);
}

// ---------------------------------------------------------------------------
// Tiled bf16 WMMA GEMM:  C[M][Nc] = A[M][K] * Bt[Nc][K]^T  (+bias, relu)
//   256 threads = 8 waves; tile 128x128x32; wave tile 32x64 (2x4 WMMA frags)
//   Double-buffered LDS, one barrier per K-step; compile-time epilogue mode.
// ---------------------------------------------------------------------------
#define BM  128
#define BN  128
#define BK  32
#define LDT 40   // LDS row stride (bf16 elems): 80B, 16B aligned, spreads banks

template <bool RELU, bool HAS_BIAS, bool OUT_F32, bool OUT_BF16>
__global__ __launch_bounds__(256)
void gemm_wmma_bf16(const unsigned short* __restrict__ A,
                    const unsigned short* __restrict__ Bt,
                    float* __restrict__ Cf,
                    unsigned short* __restrict__ Cb,
                    const float* __restrict__ bias,
                    int K, int Nc, int mtiles)
{
    __shared__ unsigned short lsA[2][BM * LDT];
    __shared__ unsigned short lsB[2][BN * LDT];

    const int bx = blockIdx.x;
    const int tm = bx % mtiles;
    const int tn = bx / mtiles;
    const int m0 = tm * BM, n0 = tn * BN;

    const int t    = threadIdx.x;
    const int lane = t & 31;
    const int wid  = t >> 5;
    const int wm   = wid & 3;          // 4 waves along M (32 rows each)
    const int wn   = wid >> 2;         // 2 waves along N (64 cols each)

    // global->LDS staging: each thread moves 16 bf16 (two b128) of one row
    const int lrow = t >> 1;           // 0..127
    const int lkh  = t & 1;            // k sub-block 0 / 16

    // fragment addressing (per ISA 16-bit A/B VGPR layouts)
    const int fr = lane & 15;
    const int kh = lane >> 4;          // 0: lanes 0-15, 1: lanes 16-31

    const unsigned short* gA = A  + (size_t)(m0 + lrow) * K + lkh * 16;
    const unsigned short* gB = Bt + (size_t)(n0 + lrow) * K + lkh * 16;
    unsigned short* la = &lsA[0][lrow * LDT + lkh * 16];
    unsigned short* lb = &lsB[0][lrow * LDT + lkh * 16];
    const int BUFSTR = BM * LDT;       // elems per LDS buffer

    const int ktiles = K / BK;

    v8f acc[2][4];
#pragma unroll
    for (int i = 0; i < 2; ++i)
#pragma unroll
        for (int j = 0; j < 4; ++j)
            acc[i][j] = (v8f){0.f,0.f,0.f,0.f,0.f,0.f,0.f,0.f};

#if USE_ASYNC_LDS
    // ---- prologue: async-stage tile 0 into buffer 0 ----
    async_b128(gA,     la);
    async_b128(gA + 8, la + 8);
    async_b128(gB,     lb);
    async_b128(gB + 8, lb + 8);
    wait_async0();
    __syncthreads();
#else
    // ---- prologue: register-stage tile 0 into buffer 0 ----
    uint4 ra0, ra1, rb0, rb1;
    {
        const uint4* sa = (const uint4*)gA;
        const uint4* sb = (const uint4*)gB;
        ra0 = sa[0]; ra1 = sa[1]; rb0 = sb[0]; rb1 = sb[1];
        uint4* da = (uint4*)la;  uint4* db = (uint4*)lb;
        da[0] = ra0; da[1] = ra1; db[0] = rb0; db[1] = rb1;
    }
    __syncthreads();
#endif

    for (int kt = 0; kt < ktiles; ++kt) {
        const int cur = kt & 1;
        const int nxt = cur ^ 1;
        const bool more = (kt + 1) < ktiles;

#if USE_ASYNC_LDS
        if (more) {   // uniform branch; EXEC untouched
            const unsigned short* sa = gA + (kt + 1) * BK;
            const unsigned short* sb = gB + (kt + 1) * BK;
            async_b128(sa,     la + nxt * BUFSTR);
            async_b128(sa + 8, la + nxt * BUFSTR + 8);
            async_b128(sb,     lb + nxt * BUFSTR);
            async_b128(sb + 8, lb + nxt * BUFSTR + 8);
        }
#else
        if (more) {   // prefetch next tile into registers while computing
            const uint4* sa = (const uint4*)(gA + (kt + 1) * BK);
            const uint4* sb = (const uint4*)(gB + (kt + 1) * BK);
            ra0 = sa[0]; ra1 = sa[1]; rb0 = sb[0]; rb1 = sb[1];
        }
#endif

        const unsigned short* baseA = &lsA[cur][0];
        const unsigned short* baseB = &lsB[cur][0];

        // A frags: lanes 0-15 (row M=fr) K 0..7 & 16..23; lanes 16-31 K 8..15 & 24..31
        v16bf afrag[2];
#pragma unroll
        for (int i = 0; i < 2; ++i) {
            const unsigned short* p = baseA + (wm * 32 + i * 16 + fr) * LDT + kh * 8;
            union { v16bf v; uint4 q[2]; } u;
            u.q[0] = *(const uint4*)(p);        // K = kh*8       -> VGPR0-3
            u.q[1] = *(const uint4*)(p + 16);   // K = 16 + kh*8  -> VGPR4-7
            afrag[i] = u.v;
        }
        // B frags: lanes 0-15 (col N=fr) K 0..7 & 16..23; lanes 16-31 K 8..15 & 24..31
        v16bf bfrag[4];
#pragma unroll
        for (int j = 0; j < 4; ++j) {
            const unsigned short* p = baseB + (wn * 64 + j * 16 + fr) * LDT + kh * 8;
            union { v16bf v; uint4 q[2]; } u;
            u.q[0] = *(const uint4*)(p);        // K = kh*8       -> VGPR0-3
            u.q[1] = *(const uint4*)(p + 16);   // K = 16 + kh*8  -> VGPR4-7
            bfrag[j] = u.v;
        }

#pragma unroll
        for (int i = 0; i < 2; ++i)
#pragma unroll
            for (int j = 0; j < 4; ++j)
                acc[i][j] = __builtin_amdgcn_wmma_f32_16x16x32_bf16(
                    false, afrag[i], false, bfrag[j],
                    (short)0, acc[i][j], false, false);

        if (more) {
#if USE_ASYNC_LDS
            wait_async0();                      // tile kt+1 resident in LDS
#else
            uint4* da = (uint4*)(la + nxt * BUFSTR);
            uint4* db = (uint4*)(lb + nxt * BUFSTR);
            da[0] = ra0; da[1] = ra1; db[0] = rb0; db[1] = rb1;
#endif
            __syncthreads();
        }
    }

    // ---- epilogue: C VGPR r holds (M = r + 8*hi, N = lane&15) per 16x16 frag
    const int cn = lane & 15;
    const int hi = lane >> 4;
#pragma unroll
    for (int i = 0; i < 2; ++i)
#pragma unroll
        for (int j = 0; j < 4; ++j) {
            const int col = n0 + wn * 64 + j * 16 + cn;
            const float bv = HAS_BIAS ? bias[col] : 0.0f;
#pragma unroll
            for (int r = 0; r < 8; ++r) {
                const int row = m0 + wm * 32 + i * 16 + r + 8 * hi;
                float v = acc[i][j][r] + bv;
                if (RELU) v = fmaxf(v, 0.0f);
                const size_t off = (size_t)row * Nc + col;
                if (OUT_F32)  Cf[off] = v;
                if (OUT_BF16) Cb[off] = f2bf(v);
            }
        }
}

// ---------------------------------------------------------------------------
// Graph message passing
// ---------------------------------------------------------------------------
__global__ void deg_kernel(const int* __restrict__ ei, float* __restrict__ deg) {
    int e = blockIdx.x * blockDim.x + threadIdx.x;
    if (e < EE) atomicAdd(&deg[ei[EE + e]], 1.0f);
}

__global__ void dinv_kernel(const float* __restrict__ deg, float* __restrict__ dinv) {
    int i = blockIdx.x * blockDim.x + threadIdx.x;
    if (i < NN) dinv[i] = rsqrtf(deg[i] + 1.0f);
}

// one block per edge: agg[col] += t[row] * dinv[row]*dinv[col]
__global__ void scatter_kernel(const float* __restrict__ t, const int* __restrict__ ei,
                               const float* __restrict__ dinv, float* __restrict__ agg, int C) {
    int e = blockIdx.x;
    int r = ei[e], c = ei[EE + e];
    float s = dinv[r] * dinv[c];
    const float* src = t + (size_t)r * C;
    float* dst = agg + (size_t)c * C;
    for (int f = threadIdx.x; f < C; f += blockDim.x)
        atomicAdd(&dst[f], src[f] * s);
}

// h = relu(agg + t*dinv^2 + bias); pad rows forced to zero
template <bool OUT_BF16>
__global__ void finalize_kernel(const float* __restrict__ t, const float* __restrict__ agg,
                                const float* __restrict__ dinv, const float* __restrict__ bias,
                                unsigned short* __restrict__ hb, float* __restrict__ hf, int C) {
    int i = blockIdx.x * blockDim.x + threadIdx.x;
    if (i >= MPAD * C) return;
    int r = i / C, c = i - r * C;
    float v = 0.0f;
    if (r < NN) {
        float di = dinv[r];
        v = fmaxf(agg[i] + t[i] * di * di + bias[c], 0.0f);
    }
    if (OUT_BF16) hb[i] = f2bf(v);
    else          hf[i] = v;
}

// ---------------------------------------------------------------------------
// Heads: one wave per node/edge, wave32 shuffle reduction
// ---------------------------------------------------------------------------
__global__ void vhead_kernel(const float* __restrict__ h, const float* __restrict__ Wv,
                             const float* __restrict__ bv, float* __restrict__ out) {
    int node = blockIdx.x * 8 + (threadIdx.x >> 5);
    int lane = threadIdx.x & 31;
    if (node >= NN) return;
    const float* hp = h + (size_t)node * LL;
    float s = 0.0f;
    for (int k = lane; k < LL; k += 32) s += hp[k] * Wv[k];
#pragma unroll
    for (int o = 16; o; o >>= 1) s += __shfl_xor(s, o, 32);
    if (lane == 0) {
        float vr = 1.0f / (1.0f + __expf(-(s + bv[0])));
        float sq = VLO + (VUP - VLO) * vr;
        out[OFF_VW + node] = sq * sq;   // vw_unp
        out[OFF_SV + node] = sq;        // sqrt(v_un)
    }
}

__global__ void ehead_kernel(const float* __restrict__ h, const int* __restrict__ ei,
                             const float* __restrict__ Wsw, const float* __restrict__ bsw,
                             float* __restrict__ out) {
    int e = blockIdx.x * 8 + (threadIdx.x >> 5);
    int lane = threadIdx.x & 31;
    if (e >= EE) return;
    int r = ei[e], c = ei[EE + e];
    const float* hr = h + (size_t)r * LL;
    const float* hc = h + (size_t)c * LL;
    float s = 0.0f;
    for (int k = lane; k < LL; k += 32)
        s += hr[k] * Wsw[k] + hc[k] * Wsw[LL + k];
#pragma unroll
    for (int o = 16; o; o >>= 1) s += __shfl_xor(s, o, 32);
    if (lane == 0) {
        float y = 1.0f / (1.0f + __expf(-(s + bsw[0])));
        out[OFF_YW + e] = y;            // yw_unp
        out[OFF_YU + e] = y;            // y_un (identity through eye matmuls)
    }
}

// ---------------------------------------------------------------------------
// Host orchestration
// ---------------------------------------------------------------------------
extern "C" void kernel_launch(void* const* d_in, const int* in_sizes, int n_in,
                              void* d_out, int out_size, void* d_ws, size_t ws_size,
                              hipStream_t stream) {
    const float* x     = (const float*)d_in[0];
    const int*   ei    = (const int*)  d_in[1];
    const float* W_enc = (const float*)d_in[15];
    const float* b_enc = (const float*)d_in[16];
    const float* W_g0  = (const float*)d_in[17];
    const float* b_g0  = (const float*)d_in[18];
    const float* W_g1  = (const float*)d_in[19];
    const float* b_g1  = (const float*)d_in[20];
    const float* W_g2  = (const float*)d_in[21];
    const float* b_g2  = (const float*)d_in[22];
    const float* W_sw  = (const float*)d_in[23];
    const float* b_sw  = (const float*)d_in[24];
    const float* W_v   = (const float*)d_in[25];
    const float* b_v   = (const float*)d_in[26];
    float* out = (float*)d_out;

    // carve workspace
    char* ws = (char*)d_ws;
    auto carve = [&](size_t bytes) -> char* {
        char* p = ws; ws += (bytes + 255) & ~(size_t)255; return p;
    };
    unsigned short* xb    = (unsigned short*)carve((size_t)MPAD * DIN * 2);
    unsigned short* wencT = (unsigned short*)carve((size_t)HH * DIN * 2);
    unsigned short* wg0T  = (unsigned short*)carve((size_t)HH * HH * 2);
    unsigned short* wg1T  = (unsigned short*)carve((size_t)HH * HH * 2);
    unsigned short* wg2T  = (unsigned short*)carve((size_t)LL * HH * 2);
    unsigned short* actA  = (unsigned short*)carve((size_t)MPAD * HH * 2);
    unsigned short* actB  = (unsigned short*)carve((size_t)MPAD * HH * 2);
    float* tbuf = (float*)carve((size_t)MPAD * HH * 4);
    float* agg  = (float*)carve((size_t)MPAD * HH * 4);
    float* h2f  = (float*)carve((size_t)MPAD * LL * 4);
    float* deg  = (float*)carve((size_t)NN * 4);
    float* dinv = (float*)carve((size_t)NN * 4);

    const int T = 256;
    // ---- precompute: casts, transposed bf16 weights, degrees -------------
    cast_x_kernel<<<(MPAD * DIN + T - 1) / T, T, 0, stream>>>(x, xb);
    castT_w_kernel<<<(DIN * HH + T - 1) / T, T, 0, stream>>>(W_enc, wencT, DIN, HH);
    castT_w_kernel<<<(HH * HH + T - 1) / T, T, 0, stream>>>(W_g0, wg0T, HH, HH);
    castT_w_kernel<<<(HH * HH + T - 1) / T, T, 0, stream>>>(W_g1, wg1T, HH, HH);
    castT_w_kernel<<<(HH * LL + T - 1) / T, T, 0, stream>>>(W_g2, wg2T, HH, LL);
    (void)hipMemsetAsync(deg, 0, (size_t)NN * 4, stream);
    deg_kernel<<<(EE + T - 1) / T, T, 0, stream>>>(ei, deg);
    dinv_kernel<<<(NN + T - 1) / T, T, 0, stream>>>(deg, dinv);

    const int MT = MPAD / BM;   // 28 M-tiles

    // ---- encoder: actA = relu(x @ W_enc + b_enc)  [bf16 out, bias+relu] --
    gemm_wmma_bf16<true, true, false, true><<<MT * (HH / BN), T, 0, stream>>>(
        xb, wencT, nullptr, actA, b_enc, DIN, HH, MT);

    // ---- GCN layer 0 -----------------------------------------------------
    gemm_wmma_bf16<false, false, true, false><<<MT * (HH / BN), T, 0, stream>>>(
        actA, wg0T, tbuf, nullptr, nullptr, HH, HH, MT);
    (void)hipMemsetAsync(agg, 0, (size_t)MPAD * HH * 4, stream);
    scatter_kernel<<<EE, T, 0, stream>>>(tbuf, ei, dinv, agg, HH);
    finalize_kernel<true><<<(MPAD * HH + T - 1) / T, T, 0, stream>>>(
        tbuf, agg, dinv, b_g0, actB, nullptr, HH);

    // ---- GCN layer 1 -----------------------------------------------------
    gemm_wmma_bf16<false, false, true, false><<<MT * (HH / BN), T, 0, stream>>>(
        actB, wg1T, tbuf, nullptr, nullptr, HH, HH, MT);
    (void)hipMemsetAsync(agg, 0, (size_t)MPAD * HH * 4, stream);
    scatter_kernel<<<EE, T, 0, stream>>>(tbuf, ei, dinv, agg, HH);
    finalize_kernel<true><<<(MPAD * HH + T - 1) / T, T, 0, stream>>>(
        tbuf, agg, dinv, b_g1, actA, nullptr, HH);

    // ---- GCN layer 2 (H -> L) -------------------------------------------
    gemm_wmma_bf16<false, false, true, false><<<MT * (LL / BN), T, 0, stream>>>(
        actA, wg2T, tbuf, nullptr, nullptr, HH, LL, MT);
    (void)hipMemsetAsync(agg, 0, (size_t)MPAD * LL * 4, stream);
    scatter_kernel<<<EE, T, 0, stream>>>(tbuf, ei, dinv, agg, LL);
    finalize_kernel<false><<<(MPAD * LL + T - 1) / T, T, 0, stream>>>(
        tbuf, agg, dinv, b_g2, nullptr, h2f, LL);

    // ---- heads -----------------------------------------------------------
    vhead_kernel<<<(NN + 7) / 8, T, 0, stream>>>(h2f, W_v, b_v, out);
    ehead_kernel<<<(EE + 7) / 8, T, 0, stream>>>(h2f, ei, W_sw, b_sw, out);
}
